// WindowAttention_16303695856204
// MI455X (gfx1250) — matrix operand order repaired and checked by reading, not compiled, not run on hardware
//
#include <hip/hip_runtime.h>

// ---------------------------------------------------------------------------
// Fused window attention for MI455X (gfx1250, wave32, WMMA bf16 16x16x32).
// One workgroup (256 threads = 8 waves) per window; all intermediates in LDS.
// Mask slice staged via GLOBAL_LOAD_ASYNC_TO_LDS (ASYNCcnt path).
// ---------------------------------------------------------------------------

typedef float  v8f  __attribute__((ext_vector_type(8)));
typedef __bf16 v16bf __attribute__((ext_vector_type(16)));

struct alignas(16) U4 { unsigned int x, y, z, w; };

union Frag {
  unsigned int u[8];
  U4           q2[2];
  v16bf        b;
};

#define NW   49
#define CDIM 128

// dword offsets inside dynamic LDS
#define OFF_X    0      // 4096 dw : x bf16 [64][128]; reused as attn scratch (8 waves * 512 dw)
#define OFF_MASK 4096   // 2404 dw : mask f32 [49*49]
#define OFF_Q    6500   // 4096 dw : q bf16 [4][64][32] (pre-scaled)
#define OFF_K    10596  // 4096 dw : k bf16 [4][64][32]
#define OFF_V    14692  // 4096 dw : v bf16 transposed [4][32][64]
#define OFF_O    18788  // 4096 dw : attn-out bf16 [64][128]
#define SMEM_DW  22884

__device__ __forceinline__ unsigned int bf1(float a) {
  unsigned int ua = __float_as_uint(a);
  return (ua + 0x7FFFu + ((ua >> 16) & 1u)) >> 16;   // round-to-nearest-even
}
__device__ __forceinline__ unsigned int pkbf(float a, float b) {
  return bf1(a) | (bf1(b) << 16);
}

__device__ __forceinline__ v8f wmma_bf16(const Frag& a, const Frag& b, v8f c) {
  return __builtin_amdgcn_wmma_f32_16x16x32_bf16(
      /*neg_a=*/false, a.b, /*neg_b=*/false, b.b,
      /*c_mod=*/(short)0, c, /*reuse_a=*/false, /*reuse_b=*/false);
}

__global__ __launch_bounds__(256) void win_attn_kernel(
    const float* __restrict__ x, const float* __restrict__ mask,
    const float* __restrict__ qkv_w, const float* __restrict__ qkv_b,
    const float* __restrict__ proj_w, const float* __restrict__ proj_b,
    float* __restrict__ out)
{
  extern __shared__ unsigned int smem[];
  unsigned short* smem16 = (unsigned short*)smem;
  float*          smemf  = (float*)smem;

  const int tid  = threadIdx.x;
  const int wave = tid >> 5;
  const int lane = tid & 31;
  const int lo   = lane & 15;     // D-layout: column / A-layout: row
  const int hi   = lane >> 4;     // half-wave selector
  const int bwin = blockIdx.x;

  // ---- stage 0a: mask slice -> LDS f32 via async-to-LDS DMA (ASYNCcnt) ----
  const float* mk = mask + (size_t)(bwin & 63) * NW * NW;
  for (int i = tid; i < NW * NW; i += 256) {
    unsigned int lds_b = (unsigned int)(OFF_MASK + i) * 4u;  // LDS byte addr
    unsigned int g_b   = (unsigned int)i * 4u;               // byte offset vs saddr
    asm volatile("global_load_async_to_lds_b32 %0, %1, %2"
                 :: "v"(lds_b), "v"(g_b), "s"(mk) : "memory");
  }

  // ---- stage 0b: x -> LDS bf16 (rows 49..63 zero) ----
  const float* xw = x + (size_t)bwin * NW * CDIM;
  for (int p = tid; p < 4096; p += 256) {
    int e = p << 1, row = e >> 7, col = e & 127;
    unsigned int v = 0u;
    if (row < NW) {
      const float* s = xw + row * CDIM + col;
      v = pkbf(s[0], s[1]);
    }
    smem[OFF_X + p] = v;
  }
  asm volatile("s_wait_asynccnt 0x0" ::: "memory");
  __syncthreads();

  // ---- stage 1: qkv = x @ qkv_w^T + b  ->  q(scaled)/k/v in LDS ----
  const float scale = 0.1767766952966369f;   // 32^-0.5
  for (int t = 0; t < 3; ++t) {
    int nt = wave * 3 + t;        // 24 channel tiles of 16 over 3C=384
    int cp = nt * 16 + lo;        // this lane's output channel
    Frag bw[4];
    const float* wr = qkv_w + (size_t)cp * CDIM;   // qkv_w[c'][k], pairs along k
#pragma unroll
    for (int kb = 0; kb < 4; ++kb)
#pragma unroll
      for (int j = 0; j < 8; ++j) {
        int k = kb * 32 + hi * 16 + 2 * j;
        bw[kb].u[j] = pkbf(wr[k], wr[k + 1]);
      }
    float bias = qkv_b[cp];
    int hh = (cp & 127) >> 5, ch = cp & 31;
    // section (q/k/v) is identical for all lanes of the wave -> scalarize
    int secu = __builtin_amdgcn_readfirstlane(cp >> 7);
    for (int m = 0; m < 4; ++m) {
      v8f acc = {};
#pragma unroll
      for (int kb = 0; kb < 4; ++kb) {
        Frag a;
        int d = OFF_X + (m * 16 + lo) * 64 + kb * 16 + hi * 8;
        a.q2[0] = *(const U4*)&smem[d];
        a.q2[1] = *(const U4*)&smem[d + 4];
        acc = wmma_bf16(a, bw[kb], acc);
      }
      if (secu == 0) {
#pragma unroll
        for (int r = 0; r < 8; ++r) {
          int row = m * 16 + r + 8 * hi;
          smem16[(OFF_Q << 1) + hh * 2048 + row * 32 + ch] =
              (unsigned short)bf1((acc[r] + bias) * scale);
        }
      } else if (secu == 1) {
#pragma unroll
        for (int r = 0; r < 8; ++r) {
          int row = m * 16 + r + 8 * hi;
          smem16[(OFF_K << 1) + hh * 2048 + row * 32 + ch] =
              (unsigned short)bf1(acc[r] + bias);
        }
      } else {                         // v transposed [h][ch][token]
        int base = hh * 2048 + ch * 64 + m * 16 + 8 * hi;
#pragma unroll
        for (int i = 0; i < 4; ++i)
          smem[OFF_V + (base >> 1) + i] =
              pkbf(acc[2 * i] + bias, acc[2 * i + 1] + bias);
      }
    }
  }
  __syncthreads();

  // ---- stage 2: per (head, m-tile): attn = softmax(qk^T + mask); o = attn@v ----
  for (int it = 0; it < 2; ++it) {
    int p  = wave + it * 8;            // 16 (head, m) pairs over 8 waves x 2
    int hh = p >> 2, m = p & 3;

    Frag aq;                           // q A-fragment (K = Dh = 32, one block)
    {
      int d = OFF_Q + hh * 1024 + (m * 16 + lo) * 16 + hi * 8;
      aq.q2[0] = *(const U4*)&smem[d];
      aq.q2[1] = *(const U4*)&smem[d + 4];
    }
    v8f at[4];
#pragma unroll
    for (int nt = 0; nt < 4; ++nt) {
      Frag bk;                         // k^T B-fragment: contiguous dwords
      int d = OFF_K + hh * 1024 + (nt * 16 + lo) * 16 + hi * 8;
      bk.q2[0] = *(const U4*)&smem[d];
      bk.q2[1] = *(const U4*)&smem[d + 4];
      v8f z = {};
      at[nt] = wmma_bf16(aq, bk, z);
    }
    // mask add + numerically-stable softmax (row lives across 16 lanes)
#pragma unroll
    for (int r = 0; r < 8; ++r) {
      int row  = m * 16 + r + 8 * hi;
      int mrow = row < NW ? row : 0;
      float mx = -3.0e38f;
#pragma unroll
      for (int nt = 0; nt < 4; ++nt) {
        int col = nt * 16 + lo;
        float v = at[nt][r];
        v = (col < NW) ? (v + smemf[OFF_MASK + mrow * NW + col]) : -3.0e38f;
        at[nt][r] = v;
        mx = fmaxf(mx, v);
      }
      mx = fmaxf(mx, __shfl_xor(mx, 1));
      mx = fmaxf(mx, __shfl_xor(mx, 2));
      mx = fmaxf(mx, __shfl_xor(mx, 4));
      mx = fmaxf(mx, __shfl_xor(mx, 8));
      float s = 0.f;
#pragma unroll
      for (int nt = 0; nt < 4; ++nt) {
        float e = __expf(at[nt][r] - mx);
        at[nt][r] = e;
        s += e;
      }
      s += __shfl_xor(s, 1);
      s += __shfl_xor(s, 2);
      s += __shfl_xor(s, 4);
      s += __shfl_xor(s, 8);
      float inv = __builtin_amdgcn_rcpf(s);
#pragma unroll
      for (int nt = 0; nt < 4; ++nt) at[nt][r] *= inv;
    }
    // P (D-layout, column-per-lane) -> per-wave LDS scratch [16][64] bf16
    int sb = (OFF_X + wave * 512) << 1;
#pragma unroll
    for (int nt = 0; nt < 4; ++nt)
#pragma unroll
      for (int r = 0; r < 8; ++r)
        smem16[sb + (r + 8 * hi) * 64 + nt * 16 + lo] =
            (unsigned short)bf1(at[nt][r]);
    __syncthreads();

    // o = P @ v  (K = 64 tokens = 2 blocks; N = Dh = 32 = 2 tiles)
    v8f oc0 = {}, oc1 = {};
#pragma unroll
    for (int kb = 0; kb < 2; ++kb) {
      Frag aa;                         // P A-fragment from scratch
      int d = OFF_X + wave * 512 + lo * 32 + kb * 16 + hi * 8;
      aa.q2[0] = *(const U4*)&smem[d];
      aa.q2[1] = *(const U4*)&smem[d + 4];
#pragma unroll
      for (int ct = 0; ct < 2; ++ct) {
        Frag bv;                       // v^T-stored => contiguous B-fragment
        int dv = OFF_V + hh * 1024 + (ct * 16 + lo) * 32 + kb * 16 + hi * 8;
        bv.q2[0] = *(const U4*)&smem[dv];
        bv.q2[1] = *(const U4*)&smem[dv + 4];
        if (ct == 0) oc0 = wmma_bf16(aa, bv, oc0);
        else         oc1 = wmma_bf16(aa, bv, oc1);
      }
    }
#pragma unroll
    for (int r = 0; r < 8; ++r) {
      int row = m * 16 + r + 8 * hi;
      smem16[(OFF_O << 1) + row * 128 + hh * 32 + lo]      = (unsigned short)bf1(oc0[r]);
      smem16[(OFF_O << 1) + row * 128 + hh * 32 + 16 + lo] = (unsigned short)bf1(oc1[r]);
    }
    __syncthreads();
  }

  // ---- stage 3: out = o @ proj_w^T + proj_b ----
  int m3 = wave >> 1;
  float* outw = out + (size_t)bwin * NW * CDIM;
  for (int t = 0; t < 4; ++t) {
    int nt = (wave & 1) * 4 + t;       // 8 channel tiles over C=128
    int cp = nt * 16 + lo;
    Frag bw[4];
    const float* wr = proj_w + (size_t)cp * CDIM;
#pragma unroll
    for (int kb = 0; kb < 4; ++kb)
#pragma unroll
      for (int j = 0; j < 8; ++j) {
        int k = kb * 32 + hi * 16 + 2 * j;
        bw[kb].u[j] = pkbf(wr[k], wr[k + 1]);
      }
    v8f acc = {};
#pragma unroll
    for (int kb = 0; kb < 4; ++kb) {
      Frag a;
      int d = OFF_O + (m3 * 16 + lo) * 64 + kb * 16 + hi * 8;
      a.q2[0] = *(const U4*)&smem[d];
      a.q2[1] = *(const U4*)&smem[d + 4];
      acc = wmma_bf16(a, bw[kb], acc);
    }
    float bias = proj_b[cp];
#pragma unroll
    for (int r = 0; r < 8; ++r) {
      int row = m3 * 16 + r + 8 * hi;
      if (row < NW) outw[row * CDIM + cp] = acc[r] + bias;
    }
  }
}

extern "C" void kernel_launch(void* const* d_in, const int* in_sizes, int n_in,
                              void* d_out, int out_size, void* d_ws, size_t ws_size,
                              hipStream_t stream) {
  const float* x      = (const float*)d_in[0];
  const float* mask   = (const float*)d_in[1];
  const float* qkv_w  = (const float*)d_in[2];
  const float* qkv_b  = (const float*)d_in[3];
  const float* proj_w = (const float*)d_in[4];
  const float* proj_b = (const float*)d_in[5];
  float* out = (float*)d_out;

  int nwin = in_sizes[0] / (NW * CDIM);   // 2048
  dim3 grid(nwin), block(256);
  size_t smem_bytes = (size_t)SMEM_DW * sizeof(unsigned int);  // ~89.4 KB
  hipLaunchKernelGGL(win_attn_kernel, grid, block, smem_bytes, stream,
                     x, mask, qkv_w, qkv_b, proj_w, proj_b, out);
}
